// HypergraphGPSLayer_9466107920684
// MI455X (gfx1250) — matrix operand order repaired and checked by reading, not compiled, not run on hardware
//
#include <hip/hip_runtime.h>
#include <math.h>

// ---------------------------------------------------------------------------
// Types for CDNA5 WMMA (wave32, bf16 16x16x32, fp32 accumulate)
// ---------------------------------------------------------------------------
typedef __attribute__((ext_vector_type(16))) __bf16 v16bf;
typedef __attribute__((ext_vector_type(8)))  __bf16 v8bf;
typedef __attribute__((ext_vector_type(8)))  float  v8f;

#define NN   16384   // nodes
#define EE   8192    // hyperedges
#define DIMC 256     // feature dim
#define HID  512     // FFN hidden
#define KC   64      // K-chunk per LDS stage
#define LDA  72      // padded LDS row (bf16 elements) for A tile (64 x KC)
#define LDB  72      // padded LDS row for B tile (256 x KC)

// ---------------------------------------------------------------------------
// WMMA fragment loaders (per CDNA5 ISA 7.12.2 layouts).
// A (16-bit, 16x32):  lane<16 -> M=lane, K = half*8+{0..7} and 16+half*8+{0..7}
// B (16-bit, 32x16):  lane -> N=lane&15, K = half*8+{0..7} and 16+half*8+{0..7}
// ---------------------------------------------------------------------------
static __device__ __forceinline__ v16bf frag_a(const __bf16* sA, int mw, int kk, int lane) {
  const int half = lane >> 4, l = lane & 15;
  const __bf16* p = sA + (size_t)(mw + l) * LDA + kk * 32 + half * 8;
  union { v16bf v; v8bf h[2]; } u;
  u.h[0] = *(const v8bf*)(p);        // K = kk*32 + half*8 + 0..7
  u.h[1] = *(const v8bf*)(p + 16);   // K = kk*32 + 16 + half*8 + 0..7
  return u.v;
}
static __device__ __forceinline__ v16bf frag_b(const __bf16* sB, int nt, int kk, int lane) {
  const int half = lane >> 4, l = lane & 15;
  const __bf16* p = sB + (size_t)(nt + l) * LDB + kk * 32 + half * 8;
  union { v16bf v; v8bf h[2]; } u;
  u.h[0] = *(const v8bf*)(p);
  u.h[1] = *(const v8bf*)(p + 16);
  return u.v;
}

static __device__ __forceinline__ float gelu_exact(float x) {
  return 0.5f * x * (1.0f + erff(x * 0.70710678118654752f));
}

// ---------------------------------------------------------------------------
// Fused degree kernel: dvsum[n] += sum_e H[n][e], desum[e] += sum_n H[n][e]
// Block: 256 thr, tile = 512 rows x 256 cols. Wave-shuffle row reduce + atomics.
// ---------------------------------------------------------------------------
__global__ __launch_bounds__(256) void deg_kernel(const float* __restrict__ H,
                                                  float* __restrict__ dvsum,
                                                  float* __restrict__ desum) {
  __shared__ float cls[8 * 256];
  const int t = threadIdx.x, lane = t & 31, w = t >> 5;
  const int c0 = blockIdx.x * 256;
  const int r0 = blockIdx.y * 512;
  float cs[8];
#pragma unroll
  for (int q = 0; q < 8; ++q) cs[q] = 0.f;
  for (int j = 0; j < 64; ++j) {
    const int r = r0 + w + 8 * j;
    const float* p = H + (size_t)r * EE + c0 + lane * 8;
    const float4 a = ((const float4*)p)[0];
    const float4 b = ((const float4*)p)[1];
    cs[0] += a.x; cs[1] += a.y; cs[2] += a.z; cs[3] += a.w;
    cs[4] += b.x; cs[5] += b.y; cs[6] += b.z; cs[7] += b.w;
    float rs = a.x + a.y + a.z + a.w + b.x + b.y + b.z + b.w;
#pragma unroll
    for (int m = 16; m > 0; m >>= 1) rs += __shfl_xor(rs, m, 32);
    if (lane == 0) atomicAdd(&dvsum[r], rs);
  }
  float* myc = &cls[w * 256 + lane * 8];
#pragma unroll
  for (int q = 0; q < 8; ++q) myc[q] = cs[q];
  __syncthreads();
  float s = 0.f;
#pragma unroll
  for (int ww = 0; ww < 8; ++ww) s += cls[ww * 256 + t];
  atomicAdd(&desum[c0 + t], s);
}

__global__ void fin_kernel(float* p, int len) {
  const int i = blockIdx.x * 256 + threadIdx.x;
  if (i < len) p[i] = rsqrtf(fmaxf(p[i], 1.0f));
}

// ---------------------------------------------------------------------------
// Transpose-convert: in f32 [K][W] (optionally scaled per input row k) ->
// out bf16 [W][K]. LDS-tiled 64x64 for coalesced loads AND stores.
// ---------------------------------------------------------------------------
__global__ __launch_bounds__(256) void tconv_kernel(const float* __restrict__ in,
                                                    const float* __restrict__ rs,
                                                    __bf16* __restrict__ out,
                                                    int W, int K) {
  __shared__ __align__(16) __bf16 tile[64 * LDA];
  const int t = threadIdx.x;
  const int k0 = blockIdx.x * 64, w0 = blockIdx.y * 64;
  {
    const int kr = t >> 2, part = t & 3;
    const float s = rs ? rs[k0 + kr] : 1.0f;
    const float* src = in + (size_t)(k0 + kr) * W + w0 + part * 16;
    float4 v[4];
#pragma unroll
    for (int i = 0; i < 4; ++i) v[i] = ((const float4*)src)[i];
    union { __bf16 h[16]; uint4 q[2]; } pk;
#pragma unroll
    for (int i = 0; i < 4; ++i) {
      pk.h[4 * i + 0] = (__bf16)(s * v[i].x);
      pk.h[4 * i + 1] = (__bf16)(s * v[i].y);
      pk.h[4 * i + 2] = (__bf16)(s * v[i].z);
      pk.h[4 * i + 3] = (__bf16)(s * v[i].w);
    }
    uint4* d = (uint4*)(tile + kr * LDA + part * 16);
    d[0] = pk.q[0]; d[1] = pk.q[1];
  }
  __syncthreads();
  {
    const int wr = t >> 2, part = t & 3;
    union { __bf16 h[16]; uint4 q[2]; } pk;
#pragma unroll
    for (int i = 0; i < 16; ++i) pk.h[i] = tile[(part * 16 + i) * LDA + wr];
    uint4* d = (uint4*)(out + (size_t)(w0 + wr) * K + k0 + part * 16);
    d[0] = pk.q[0]; d[1] = pk.q[1];
  }
}

// Row-scaled f32 -> bf16 row-major (finalize big-GEMM accumulators)
__global__ void rsconv_kernel(const float* __restrict__ in, const float* __restrict__ rs,
                              __bf16* __restrict__ out, int total) {
  const int i = blockIdx.x * 256 + threadIdx.x;
  if (i < total) out[i] = (__bf16)(rs[i >> 8] * in[i]);
}

// ---------------------------------------------------------------------------
// Big WMMA GEMM: Cacc[M][256] += op(H) @ BT^T  (fp32 atomic K-split accumulate)
//   TRANST=0: A[m][k] = H[m*EE + k]   (Hn @ X path, M=N, K=E)
//   TRANST=1: A[m][k] = H[k*EE + m]   (Hn^T @ X path, M=E, K=N)
// BT: bf16 [256][Ktot] (transposed B operand). Per-m scale applied later.
// Pipelining: only the HBM-bound A (H) stream is register-prefetched across
// the compute phase (16 VGPRs); B is L2-resident and staged after compute.
// __launch_bounds__(256, 2): cap at 512 VGPRs so nothing spills to scratch.
// ---------------------------------------------------------------------------
template <int TRANST>
__global__ __launch_bounds__(256, 2) void hg_big(const float* __restrict__ H,
                                                 const __bf16* __restrict__ BT,
                                                 float* __restrict__ Cacc,
                                                 int Ktot, int kPerSplit) {
  __shared__ __align__(16) __bf16 sA[64 * LDA];
  __shared__ __align__(16) __bf16 sB[256 * LDB];
  const int t = threadIdx.x, lane = t & 31, wid = t >> 5;
  const int m0 = blockIdx.x * 64;
  const int kBase = blockIdx.y * kPerSplit;
  const int mw = (wid >> 1) * 16;
  const int nb = (wid & 1) * 128;
  v8f acc[8];
#pragma unroll
  for (int i = 0; i < 8; ++i)
#pragma unroll
    for (int e = 0; e < 8; ++e) acc[i][e] = 0.f;

  const int rA = t >> 2, pA = t & 3;
  float4 va[4];  // persistent A-prefetch registers (the HBM stream)

  auto load_a = [&](int kc0) {
    const float* srcA = TRANST
        ? (H + (size_t)(kc0 + rA) * EE + (m0 + pA * 16))
        : (H + (size_t)(m0 + rA) * EE + (kc0 + pA * 16));
#pragma unroll
    for (int i = 0; i < 4; ++i) va[i] = ((const float4*)srcA)[i];
  };
  auto store_a = [&]() {
    if (TRANST) {
      // rA is the k index, pA*16.. are 16 consecutive m values -> scatter rows
      __bf16* dst = sA + (pA * 16) * LDA + rA;
#pragma unroll
      for (int i = 0; i < 4; ++i) {
        dst[(4 * i + 0) * LDA] = (__bf16)va[i].x;
        dst[(4 * i + 1) * LDA] = (__bf16)va[i].y;
        dst[(4 * i + 2) * LDA] = (__bf16)va[i].z;
        dst[(4 * i + 3) * LDA] = (__bf16)va[i].w;
      }
    } else {
      union { __bf16 h[16]; uint4 q[2]; } pk;
#pragma unroll
      for (int i = 0; i < 4; ++i) {
        pk.h[4 * i + 0] = (__bf16)va[i].x;
        pk.h[4 * i + 1] = (__bf16)va[i].y;
        pk.h[4 * i + 2] = (__bf16)va[i].z;
        pk.h[4 * i + 3] = (__bf16)va[i].w;
      }
      uint4* d = (uint4*)(sA + rA * LDA + pA * 16);
      d[0] = pk.q[0]; d[1] = pk.q[1];
    }
  };
  auto stage_b = [&](int kc0) {  // transient regs; B hits L2 (X fits easily)
    const uint4* srcB = (const uint4*)(BT + (size_t)t * Ktot + kc0);
    uint4 qb[8];
#pragma unroll
    for (int i = 0; i < 8; ++i) qb[i] = srcB[i];
#pragma unroll
    for (int i = 0; i < 8; ++i) *((uint4*)(sB + t * LDB + i * 8)) = qb[i];
  };

  const int chunks = kPerSplit / KC;
  load_a(kBase);
  store_a();
  stage_b(kBase);
  for (int c = 0; c < chunks; ++c) {
    __syncthreads();  // staged chunk c visible
    if (c + 1 < chunks) load_a(kBase + (c + 1) * KC);  // HBM prefetch in flight
#pragma unroll
    for (int kk = 0; kk < 2; ++kk) {
      const v16bf a = frag_a(sA, mw, kk, lane);
#pragma unroll
      for (int tt = 0; tt < 8; ++tt) {
        const v16bf b = frag_b(sB, nb + tt * 16, kk, lane);
        acc[tt] = __builtin_amdgcn_wmma_f32_16x16x32_bf16(
            false, a, false, b, (short)0, acc[tt], false, false);
      }
    }
    if (c + 1 < chunks) {
      __syncthreads();                     // all reads of chunk c done
      store_a();                           // load-waits here, after compute
      stage_b(kBase + (c + 1) * KC);
    }
  }
  // atomic accumulate (K-split partials)
  const int half = lane >> 4, l15 = lane & 15;
#pragma unroll
  for (int tt = 0; tt < 8; ++tt)
#pragma unroll
    for (int j = 0; j < 8; ++j) {
      const int m = m0 + mw + half * 8 + j;
      const int n = nb + tt * 16 + l15;
      atomicAdd(&Cacc[(size_t)m * DIMC + n], acc[tt][j]);
    }
}

// ---------------------------------------------------------------------------
// Dense WMMA GEMM: out = epilogue(A[M][K] @ WT^T + bias)
//   A: bf16 row-major. WT: bf16 [Nw][K] (transposed weights).
//   mode 0: outf = resid + (acc+bias)
//   mode 1: outf = resid + tanh(gate)*(acc+bias)
//   mode 2: outb = bf16(gelu(acc+bias))
// gridDim.y tiles the output N dimension in steps of 256 (ldout = total Nw).
// ---------------------------------------------------------------------------
__global__ __launch_bounds__(256, 2) void hg_dense(const __bf16* __restrict__ A,
                                                   const __bf16* __restrict__ WT,
                                                   int K, int ldout, int mode,
                                                   const float* __restrict__ bias,
                                                   const float* __restrict__ resid,
                                                   const float* __restrict__ gate,
                                                   float* __restrict__ outf,
                                                   __bf16* __restrict__ outb) {
  __shared__ __align__(16) __bf16 sA[64 * LDA];
  __shared__ __align__(16) __bf16 sB[256 * LDB];
  const int t = threadIdx.x, lane = t & 31, wid = t >> 5;
  const int m0 = blockIdx.x * 64;
  const int nOff = blockIdx.y * 256;
  const int mw = (wid >> 1) * 16;
  const int nb = (wid & 1) * 128;
  v8f acc[8];
#pragma unroll
  for (int i = 0; i < 8; ++i)
#pragma unroll
    for (int e = 0; e < 8; ++e) acc[i][e] = 0.f;

  const int chunks = K / KC;
  for (int c = 0; c < chunks; ++c) {
    const int kc0 = c * KC;
    {
      const int r = t >> 2, part = t & 3;
      const uint4* s4 = (const uint4*)(A + (size_t)(m0 + r) * K + kc0 + part * 16);
      uint4* d = (uint4*)(sA + r * LDA + part * 16);
      d[0] = s4[0]; d[1] = s4[1];
    }
    {
      const uint4* s4 = (const uint4*)(WT + (size_t)(nOff + t) * K + kc0);
#pragma unroll
      for (int i = 0; i < 8; ++i) *((uint4*)(sB + t * LDB + i * 8)) = s4[i];
    }
    __syncthreads();
#pragma unroll
    for (int kk = 0; kk < 2; ++kk) {
      const v16bf a = frag_a(sA, mw, kk, lane);
#pragma unroll
      for (int tt = 0; tt < 8; ++tt) {
        const v16bf b = frag_b(sB, nb + tt * 16, kk, lane);
        acc[tt] = __builtin_amdgcn_wmma_f32_16x16x32_bf16(
            false, a, false, b, (short)0, acc[tt], false, false);
      }
    }
    __syncthreads();
  }
  const int half = lane >> 4, l15 = lane & 15;
  const float g = (mode == 1) ? tanhf(gate[0]) : 0.f;
#pragma unroll
  for (int tt = 0; tt < 8; ++tt)
#pragma unroll
    for (int j = 0; j < 8; ++j) {
      const int m = m0 + mw + half * 8 + j;
      const int ng = nOff + nb + tt * 16 + l15;
      const float val = acc[tt][j] + bias[ng];
      const size_t idx = (size_t)m * ldout + ng;
      if (mode == 0)      outf[idx] = resid[idx] + val;
      else if (mode == 1) outf[idx] = resid[idx] + g * val;
      else                outb[idx] = (__bf16)gelu_exact(val);
    }
}

// ---------------------------------------------------------------------------
// LayerNorm over 256 features, one block per row. Optional bf16 copy out.
// ---------------------------------------------------------------------------
__global__ __launch_bounds__(256) void ln_kernel(const float* __restrict__ in,
                                                 const float* __restrict__ g,
                                                 const float* __restrict__ b,
                                                 float* __restrict__ outf,
                                                 __bf16* __restrict__ outb) {
  __shared__ float red[256];
  const int row = blockIdx.x, t = threadIdx.x;
  const float x = in[(size_t)row * 256 + t];
  red[t] = x;
  __syncthreads();
  for (int s = 128; s > 0; s >>= 1) { if (t < s) red[t] += red[t + s]; __syncthreads(); }
  const float mean = red[0] * (1.0f / 256.0f);
  __syncthreads();
  const float d = x - mean;
  red[t] = d * d;
  __syncthreads();
  for (int s = 128; s > 0; s >>= 1) { if (t < s) red[t] += red[t + s]; __syncthreads(); }
  const float var = red[0] * (1.0f / 256.0f);
  const float y = d * rsqrtf(var + 1e-5f) * g[t] + b[t];
  outf[(size_t)row * 256 + t] = y;
  if (outb) outb[(size_t)row * 256 + t] = (__bf16)y;
}

// ---------------------------------------------------------------------------
// Host orchestration
// ---------------------------------------------------------------------------
extern "C" void kernel_launch(void* const* d_in, const int* in_sizes, int n_in,
                              void* d_out, int out_size, void* d_ws, size_t ws_size,
                              hipStream_t stream) {
  (void)in_sizes; (void)n_in; (void)out_size; (void)ws_size;
  const float* x0  = (const float*)d_in[0];
  const float* x1  = (const float*)d_in[1];
  const float* H   = (const float*)d_in[2];
  const float* Wn  = (const float*)d_in[3];
  const float* bn  = (const float*)d_in[4];
  const float* We  = (const float*)d_in[5];
  const float* be  = (const float*)d_in[6];
  const float* Wr  = (const float*)d_in[7];
  const float* br  = (const float*)d_in[8];
  const float* W1  = (const float*)d_in[9];
  const float* b1  = (const float*)d_in[10];
  const float* W2  = (const float*)d_in[11];
  const float* b2  = (const float*)d_in[12];
  const float* ln1g = (const float*)d_in[13];
  const float* ln1b = (const float*)d_in[14];
  const float* ln2g = (const float*)d_in[15];
  const float* ln2b = (const float*)d_in[16];
  const float* gl  = (const float*)d_in[17];
  const float* gr  = (const float*)d_in[18];

  float* out0 = (float*)d_out;                 // x_0 out [N,256]
  float* out1 = out0 + (size_t)NN * DIMC;      // x_1 out [E,256]

  char* w = (char*)d_ws;
  size_t off = 0;
  auto take = [&](size_t bytes) -> char* {
    char* r = w + off;
    off += (bytes + 255) & ~(size_t)255;
    return r;
  };
  float*  rv    = (float*) take((size_t)NN * 4);            // dvsum -> rsqrt
  float*  re    = (float*) take((size_t)EE * 4);            // desum -> rsqrt
  __bf16* bT    = (__bf16*)take((size_t)DIMC * NN * 2);     // x0bT / x1newbT / x0localbT
  float*  Cacc  = (float*) take((size_t)NN * DIMC * 4);     // big-GEMM accumulator
  float*  x1new = (float*) take((size_t)EE * DIMC * 4);     // x_1 after edge update
  __bf16* Tb    = (__bf16*)take((size_t)EE * DIMC * 2);     // T0b / Rb (dense-A operands)
  __bf16* Mb    = (__bf16*)take((size_t)NN * DIMC * 2);     // Hn@x1 bf16 / x0g bf16
  float*  pre1  = (float*) take((size_t)NN * DIMC * 4);     // pre-LN1 / x0_global f32
  float*  x0l   = (float*) take((size_t)NN * DIMC * 4);     // x0_local f32; later hidb
  __bf16* hidb  = (__bf16*)x0l;                             // FFN hidden bf16 [N,512]
  __bf16* WeT   = (__bf16*)take((size_t)DIMC * DIMC * 2);
  __bf16* WnT   = (__bf16*)take((size_t)DIMC * DIMC * 2);
  __bf16* WrT   = (__bf16*)take((size_t)DIMC * DIMC * 2);
  __bf16* W1T   = (__bf16*)take((size_t)DIMC * HID * 2);
  __bf16* W2T   = (__bf16*)take((size_t)HID * DIMC * 2);

  // ---- degrees: rv = rsqrt(max(rowdeg,1)), re = rsqrt(max(coldeg,1)) ----
  hipMemsetAsync(rv, 0, (size_t)NN * 4, stream);
  hipMemsetAsync(re, 0, (size_t)EE * 4, stream);
  deg_kernel<<<dim3(EE / 256, NN / 512), 256, 0, stream>>>(H, rv, re);
  fin_kernel<<<NN / 256, 256, 0, stream>>>(rv, NN);
  fin_kernel<<<EE / 256, 256, 0, stream>>>(re, EE);

  // ---- weight / input converts (transposed bf16) ----
  tconv_kernel<<<dim3(NN / 64, DIMC / 64), 256, 0, stream>>>(x0, rv, bT, DIMC, NN);
  tconv_kernel<<<dim3(DIMC / 64, DIMC / 64), 256, 0, stream>>>(We, nullptr, WeT, DIMC, DIMC);
  tconv_kernel<<<dim3(DIMC / 64, DIMC / 64), 256, 0, stream>>>(Wn, nullptr, WnT, DIMC, DIMC);
  tconv_kernel<<<dim3(DIMC / 64, DIMC / 64), 256, 0, stream>>>(Wr, nullptr, WrT, DIMC, DIMC);
  tconv_kernel<<<dim3(DIMC / 64, HID / 64), 256, 0, stream>>>(W1, nullptr, W1T, HID, DIMC);
  tconv_kernel<<<dim3(HID / 64, DIMC / 64), 256, 0, stream>>>(W2, nullptr, W2T, DIMC, HID);

  // ---- G1: Hn^T @ x0  -> Tb (bf16, re-scaled rows) ----
  hipMemsetAsync(Cacc, 0, (size_t)EE * DIMC * 4, stream);
  hg_big<1><<<dim3(EE / 64, 4), 256, 0, stream>>>(H, bT, Cacc, NN, NN / 4);
  rsconv_kernel<<<(EE * DIMC) / 256, 256, 0, stream>>>(Cacc, re, Tb, EE * DIMC);

  // ---- E1: x1new = x1 + Tb @ We + be ----
  hg_dense<<<dim3(EE / 64, 1), 256, 0, stream>>>(Tb, WeT, DIMC, DIMC, 0, be, x1, nullptr, x1new, nullptr);
  tconv_kernel<<<dim3(EE / 64, DIMC / 64), 256, 0, stream>>>(x1new, re, bT, DIMC, EE);

  // ---- G2: Hn @ x1new -> Mb (bf16, rv-scaled rows) ----
  hipMemsetAsync(Cacc, 0, (size_t)NN * DIMC * 4, stream);
  hg_big<0><<<dim3(NN / 64, 2), 256, 0, stream>>>(H, bT, Cacc, EE, EE / 2);
  rsconv_kernel<<<(NN * DIMC) / 256, 256, 0, stream>>>(Cacc, rv, Mb, NN * DIMC);

  // ---- E2 + LN1: x0_local ----
  hg_dense<<<dim3(NN / 64, 1), 256, 0, stream>>>(Mb, WnT, DIMC, DIMC, 1, bn, x0, gl, pre1, nullptr);
  ln_kernel<<<NN, 256, 0, stream>>>(pre1, ln1g, ln1b, x0l, nullptr);
  tconv_kernel<<<dim3(NN / 64, DIMC / 64), 256, 0, stream>>>(x0l, rv, bT, DIMC, NN);

  // ---- G3: Hn^T @ x0_local -> Tb ----
  hipMemsetAsync(Cacc, 0, (size_t)EE * DIMC * 4, stream);
  hg_big<1><<<dim3(EE / 64, 4), 256, 0, stream>>>(H, bT, Cacc, NN, NN / 4);
  rsconv_kernel<<<(EE * DIMC) / 256, 256, 0, stream>>>(Cacc, re, Tb, EE * DIMC);

  // ---- E3: x1 output = x1new + tanh(gr)*(Tb @ Wr + br) ----
  hg_dense<<<dim3(EE / 64, 1), 256, 0, stream>>>(Tb, WrT, DIMC, DIMC, 1, br, x1new, gr, out1, nullptr);

  // ---- LN2 + FFN ----
  ln_kernel<<<NN, 256, 0, stream>>>(x0l, ln2g, ln2b, pre1, Mb);  // pre1 = x0g f32, Mb = x0g bf16
  hg_dense<<<dim3(NN / 64, 2), 256, 0, stream>>>(Mb, W1T, DIMC, HID, 2, b1, nullptr, nullptr, nullptr, hidb);
  hg_dense<<<dim3(NN / 64, 1), 256, 0, stream>>>(hidb, W2T, HID, DIMC, 0, b2, pre1, nullptr, out0, nullptr);
}
